// TransformerEncoder_214748364916
// MI455X (gfx1250) — compile-verified, matrix operand order checked by reading, not run on hardware
//
#include <hip/hip_runtime.h>

typedef __bf16 bf16_t;
typedef __attribute__((ext_vector_type(16))) __bf16 v16bf;
typedef __attribute__((ext_vector_type(8)))  float  v8f;
typedef int v4i_gcc __attribute__((vector_size(16)));   // matches builtin's V4i param

#define NB 8
#define NS 1000
#define NDM 512
#define NH 8
#define NDH 64
#define NHID 2048
#define NL 6
#define S2 1024
#define MBS (NB*NS)   // 8000
#define NBH (NB*NH)   // 64

#define GAS __attribute__((address_space(1)))
#define LAS __attribute__((address_space(3)))

#if defined(__has_builtin)
#  if __has_builtin(__builtin_amdgcn_global_load_async_to_lds_b128) && \
      __has_builtin(__builtin_amdgcn_s_wait_asynccnt)
#    define USE_ASYNC_LDS 1
#  endif
#endif

union AFrag { v16bf v; uint4 u[2]; };

static __device__ __forceinline__ bf16_t f2bf(float f) {
  union { float f; unsigned u; } in; in.f = f;
  unsigned r = (in.u + 0x7FFFu + ((in.u >> 16) & 1u)) >> 16;
  union { unsigned short s; bf16_t b; } out; out.s = (unsigned short)r;
  return out.b;
}

// ---------------- embedding + positional ----------------
__global__ void embed_kernel(const int* __restrict__ tokens,
                             const float* __restrict__ emb,
                             const float* __restrict__ pos,
                             float* __restrict__ x) {
  int idx = blockIdx.x * 256 + threadIdx.x;           // 8000*512 total
  int d = idx & (NDM - 1);
  int bsi = idx >> 9;                                  // b*S + s
  int s = bsi % NS;
  int tok = tokens[bsi];
  x[idx] = emb[(size_t)tok * NDM + d] * 22.62741699796952f + pos[s * NDM + d];
}

// ---------------- f32 -> bf16 convert ----------------
__global__ void cvt_kernel(const float* __restrict__ in, bf16_t* __restrict__ out, int n) {
  int i = blockIdx.x * 256 + threadIdx.x;
  if (i < n) out[i] = f2bf(in[i]);
}

// ---------------- [B,S,H,DH] f32 -> [B,H,S,DH] bf16 ----------------
__global__ void head_reshape_kernel(const float* __restrict__ in, bf16_t* __restrict__ out) {
  int idx = blockIdx.x * 256 + threadIdx.x;           // 8000*512
  int d = idx & (NDH - 1);
  int h = (idx >> 6) & (NH - 1);
  int bsi = idx >> 9;
  int s = bsi % NS, b = bsi / NS;
  out[(((size_t)(b * NH + h)) * NS + s) * NDH + d] = f2bf(in[idx]);
}

// ---------------- V f32 [B,S,H,DH] -> Vt bf16 [B,H,DH,S2] (zero padded) ----------------
__global__ void vt_reshape_kernel(const float* __restrict__ in, bf16_t* __restrict__ out) {
  int idx = blockIdx.x * 256 + threadIdx.x;           // 64*64*1024
  int s = idx & (S2 - 1);
  int d = (idx >> 10) & (NDH - 1);
  int bh = idx >> 16;
  int h = bh & (NH - 1), b = bh >> 3;
  float v = 0.0f;
  if (s < NS) v = in[((size_t)(b * NS + s)) * NDM + h * NDH + d];
  out[idx] = f2bf(v);
}

// ---------------- generic WMMA GEMM: C[M,N] = A[M,K](bf16) * W[N,K]^T + bias ----------------
#define BM 128
#define BN 64
#define BK 32
#define BKP 40

__global__ void __launch_bounds__(256) gemm_kernel(
    const bf16_t* __restrict__ A, int lda,
    const bf16_t* __restrict__ W, int ldw,
    float* __restrict__ C, int ldc,
    int M, int N, int K,
    const float* __restrict__ bias, int relu) {
  int m0 = blockIdx.x * BM;
  int n0 = blockIdx.y * BN;
  int t = threadIdx.x;
  int w = t >> 5, l = t & 31, l15 = l & 15, hi = l >> 4;
  int wm = w & 3, wn = w >> 2;

  v8f acc00 = {}, acc01 = {}, acc10 = {}, acc11 = {};
  int nk = K / BK;

#ifdef USE_ASYNC_LDS
  // ---- triple-buffered async staging: ASYNCcnt-tracked DMA into LDS ----
  __shared__ bf16_t As[3][BM][BKP];
  __shared__ bf16_t Bs[3][BN][BKP];

  // A: 128 rows x 32 bf16 = 512 x 16B chunks -> thread t loads 32B of row t>>1
  int arow = t >> 1;
  int aoff = (t & 1) * 16;                  // bf16 element offset 0 / 16
  int ag = m0 + arow; if (ag > M - 1) ag = M - 1;
  // B: 64 rows x 4 quarters -> thread t loads 16B
  int brow = t >> 2;
  int boff = (t & 3) * 8;
  const bf16_t* agp = A + (size_t)ag * lda + aoff;
  const bf16_t* bgp = W + (size_t)(n0 + brow) * ldw + boff;

  auto issue = [&](int kk, int buf) {
    int kb = kk * BK;
    __builtin_amdgcn_global_load_async_to_lds_b128(
        (GAS v4i_gcc*)(agp + kb), (LAS v4i_gcc*)&As[buf][arow][aoff], 0, 0);
    __builtin_amdgcn_global_load_async_to_lds_b128(
        (GAS v4i_gcc*)(agp + kb + 8), (LAS v4i_gcc*)&As[buf][arow][aoff + 8], 0, 0);
    __builtin_amdgcn_global_load_async_to_lds_b128(
        (GAS v4i_gcc*)(bgp + kb), (LAS v4i_gcc*)&Bs[buf][brow][boff], 0, 0);
  };

  issue(0, 0);
  if (nk > 1) issue(1, 1);

  for (int kk = 0; kk < nk; kk++) {
    int buf = kk % 3;
    if (kk + 1 < nk) __builtin_amdgcn_s_wait_asynccnt(3);
    else             __builtin_amdgcn_s_wait_asynccnt(0);
    __syncthreads();
    if (kk + 2 < nk) issue(kk + 2, (kk + 2) % 3);

    const uint4* a_lds0 = (const uint4*)&As[buf][wm * 32 +      l15][hi * 8];
    const uint4* a_lds1 = (const uint4*)&As[buf][wm * 32 + 16 + l15][hi * 8];
    const uint4* b_lds0 = (const uint4*)&Bs[buf][wn * 32 +      l15][hi * 8];
    const uint4* b_lds1 = (const uint4*)&Bs[buf][wn * 32 + 16 + l15][hi * 8];
    AFrag a0, a1, b0, b1;
    a0.u[0] = a_lds0[0]; a0.u[1] = a_lds0[2];
    a1.u[0] = a_lds1[0]; a1.u[1] = a_lds1[2];
    b0.u[0] = b_lds0[0]; b0.u[1] = b_lds0[2];
    b1.u[0] = b_lds1[0]; b1.u[1] = b_lds1[2];
    acc00 = __builtin_amdgcn_wmma_f32_16x16x32_bf16(false, a0.v, false, b0.v, (short)0, acc00, false, false);
    acc01 = __builtin_amdgcn_wmma_f32_16x16x32_bf16(false, a0.v, false, b1.v, (short)0, acc01, false, false);
    acc10 = __builtin_amdgcn_wmma_f32_16x16x32_bf16(false, a1.v, false, b0.v, (short)0, acc10, false, false);
    acc11 = __builtin_amdgcn_wmma_f32_16x16x32_bf16(false, a1.v, false, b1.v, (short)0, acc11, false, false);
  }
#else
  // ---- fallback: synchronous staging ----
  __shared__ bf16_t As[BM][BKP];
  __shared__ bf16_t Bs[BN][BKP];
  int arow = t >> 1;
  int ahalf = (t & 1) * 16;
  int ag = m0 + arow; if (ag > M - 1) ag = M - 1;
  int brow = arow & 63;
  bool doB = (t < 128);

  const uint4* a_lds0 = (const uint4*)&As[wm * 32 +      l15][hi * 8];
  const uint4* a_lds1 = (const uint4*)&As[wm * 32 + 16 + l15][hi * 8];
  const uint4* b_lds0 = (const uint4*)&Bs[wn * 32 +      l15][hi * 8];
  const uint4* b_lds1 = (const uint4*)&Bs[wn * 32 + 16 + l15][hi * 8];

  for (int kk = 0; kk < nk; kk++) {
    int kb = kk * BK;
    *(uint4*)&As[arow][ahalf] = *(const uint4*)(A + (size_t)ag * lda + kb + ahalf);
    if (doB)
      *(uint4*)&Bs[brow][ahalf] = *(const uint4*)(W + (size_t)(n0 + brow) * ldw + kb + ahalf);
    __syncthreads();
    AFrag a0, a1, b0, b1;
    a0.u[0] = a_lds0[0]; a0.u[1] = a_lds0[2];
    a1.u[0] = a_lds1[0]; a1.u[1] = a_lds1[2];
    b0.u[0] = b_lds0[0]; b0.u[1] = b_lds0[2];
    b1.u[0] = b_lds1[0]; b1.u[1] = b_lds1[2];
    acc00 = __builtin_amdgcn_wmma_f32_16x16x32_bf16(false, a0.v, false, b0.v, (short)0, acc00, false, false);
    acc01 = __builtin_amdgcn_wmma_f32_16x16x32_bf16(false, a0.v, false, b1.v, (short)0, acc01, false, false);
    acc10 = __builtin_amdgcn_wmma_f32_16x16x32_bf16(false, a1.v, false, b0.v, (short)0, acc10, false, false);
    acc11 = __builtin_amdgcn_wmma_f32_16x16x32_bf16(false, a1.v, false, b1.v, (short)0, acc11, false, false);
    __syncthreads();
  }
#endif

  auto store_tile = [&](v8f cc, int sm, int sn) {
    int col = n0 + wn * 32 + sn * 16 + l15;
    float bv = bias ? bias[col] : 0.0f;
#pragma unroll
    for (int r = 0; r < 8; r++) {
      int row = m0 + wm * 32 + sm * 16 + hi * 8 + r;
      if (row < M) {
        float v = cc[r] + bv;
        if (relu) v = fmaxf(v, 0.0f);
        C[(size_t)row * ldc + col] = v;
      }
    }
  };
  store_tile(acc00, 0, 0);
  store_tile(acc01, 0, 1);
  store_tile(acc10, 1, 0);
  store_tile(acc11, 1, 1);
}

// ---------------- fused scores (Q K^T / 8) + mask + softmax -> bf16 probs ----------------
__global__ void __launch_bounds__(256) scores_softmax_kernel(
    const bf16_t* __restrict__ qb, const bf16_t* __restrict__ kb,
    bf16_t* __restrict__ attnb, const int* __restrict__ vlens) {
  __shared__ float sc[16][S2];   // 64 KB
  int bh = blockIdx.x;
  int b = bh >> 3;
  int q0 = blockIdx.y * 16;
  int t = threadIdx.x;
  int w = t >> 5, l = t & 31, l15 = l & 15, hi = l >> 4;

  int qrow = q0 + l15; if (qrow > NS - 1) qrow = NS - 1;
  const bf16_t* qptr = qb + ((size_t)bh * NS + qrow) * NDH;
  AFrag a0, a1;
  a0.u[0] = *(const uint4*)(qptr + hi * 8);
  a0.u[1] = *(const uint4*)(qptr + hi * 8 + 16);
  a1.u[0] = *(const uint4*)(qptr + 32 + hi * 8);
  a1.u[1] = *(const uint4*)(qptr + 32 + hi * 8 + 16);

  for (int kt = w; kt < 63; kt += 8) {
    int krow = kt * 16 + l15; if (krow > NS - 1) krow = NS - 1;
    const bf16_t* kptr = kb + ((size_t)bh * NS + krow) * NDH;
    AFrag b0, b1;
    b0.u[0] = *(const uint4*)(kptr + hi * 8);
    b0.u[1] = *(const uint4*)(kptr + hi * 8 + 16);
    b1.u[0] = *(const uint4*)(kptr + 32 + hi * 8);
    b1.u[1] = *(const uint4*)(kptr + 32 + hi * 8 + 16);
    v8f c = {};
    c = __builtin_amdgcn_wmma_f32_16x16x32_bf16(false, a0.v, false, b0.v, (short)0, c, false, false);
    c = __builtin_amdgcn_wmma_f32_16x16x32_bf16(false, a1.v, false, b1.v, (short)0, c, false, false);
#pragma unroll
    for (int r = 0; r < 8; r++)
      sc[hi * 8 + r][kt * 16 + l15] = c[r] * 0.125f;
  }
  __syncthreads();

  int vlen = vlens[b];
  for (int rr = 0; rr < 2; rr++) {
    int row = w * 2 + rr;
    float mx = -3.4e38f;
    for (int k = l; k < NS; k += 32) {
      float v = (k >= vlen) ? 1e-6f : sc[row][k];
      sc[row][k] = v;
      mx = fmaxf(mx, v);
    }
    for (int m = 16; m > 0; m >>= 1) mx = fmaxf(mx, __shfl_xor(mx, m, 32));
    float sum = 0.0f;
    for (int k = l; k < NS; k += 32) {
      float e = __expf(sc[row][k] - mx);
      sc[row][k] = e;
      sum += e;
    }
    for (int m = 16; m > 0; m >>= 1) sum += __shfl_xor(sum, m, 32);
    float inv = 1.0f / sum;
    int q = q0 + row;
    if (q < NS) {
      bf16_t* op = attnb + ((size_t)bh * NS + q) * S2;
      for (int k = l; k < S2; k += 32)
        op[k] = (k < NS) ? f2bf(sc[row][k] * inv) : f2bf(0.0f);
    }
  }
}

// ---------------- attn (bf16, padded K=1024) x V^T -> o [B,S,DM] merged heads ----------------
__global__ void __launch_bounds__(128) attnv_kernel(
    const bf16_t* __restrict__ attnb, const bf16_t* __restrict__ vtb,
    float* __restrict__ o) {
  int bh = blockIdx.x;
  int b = bh >> 3, h = bh & (NH - 1);
  int q0 = blockIdx.y * 16;
  int t = threadIdx.x;
  int w = t >> 5, l = t & 31, l15 = l & 15, hi = l >> 4;

  int qrow = q0 + l15; if (qrow > NS - 1) qrow = NS - 1;
  const bf16_t* ap = attnb + ((size_t)bh * NS + qrow) * S2 + hi * 8;
  const bf16_t* bp = vtb + ((size_t)bh * NDH + w * 16 + l15) * S2 + hi * 8;

  v8f c = {};
#pragma unroll 4
  for (int ks = 0; ks < 32; ks++) {
    AFrag a, bb;
    a.u[0]  = *(const uint4*)(ap + ks * 32);
    a.u[1]  = *(const uint4*)(ap + ks * 32 + 16);
    bb.u[0] = *(const uint4*)(bp + ks * 32);
    bb.u[1] = *(const uint4*)(bp + ks * 32 + 16);
    c = __builtin_amdgcn_wmma_f32_16x16x32_bf16(false, a.v, false, bb.v, (short)0, c, false, false);
  }
#pragma unroll
  for (int r = 0; r < 8; r++) {
    int q = q0 + hi * 8 + r;
    if (q < NS)
      o[((size_t)b * NS + q) * NDM + h * NDH + w * 16 + l15] = c[r];
  }
}

// ---------------- fused residual + LayerNorm ----------------
__global__ void __launch_bounds__(256) ln_kernel(
    const float* __restrict__ resid, const float* __restrict__ y,
    const float* __restrict__ g, const float* __restrict__ beta,
    float* __restrict__ out) {
  int row = blockIdx.x;
  int t = threadIdx.x;
  const float* r = resid + (size_t)row * NDM;
  const float* p = y + (size_t)row * NDM;
  float v0 = r[t] + p[t];
  float v1 = r[t + 256] + p[t + 256];
  float s1 = v0 + v1, s2 = v0 * v0 + v1 * v1;
  for (int m = 16; m > 0; m >>= 1) {
    s1 += __shfl_xor(s1, m, 32);
    s2 += __shfl_xor(s2, m, 32);
  }
  __shared__ float a1[8], a2[8];
  int w = t >> 5, ln = t & 31;
  if (ln == 0) { a1[w] = s1; a2[w] = s2; }
  __syncthreads();
  if (t == 0) {
    float t1 = 0, t2 = 0;
    for (int i = 0; i < 8; i++) { t1 += a1[i]; t2 += a2[i]; }
    a1[0] = t1; a2[0] = t2;
  }
  __syncthreads();
  float mu = a1[0] * (1.0f / NDM);
  float var = a2[0] * (1.0f / NDM) - mu * mu;
  float rs = rsqrtf(var + 1e-5f);
  out[(size_t)row * NDM + t]       = (v0 - mu) * rs * g[t] + beta[t];
  out[(size_t)row * NDM + t + 256] = (v1 - mu) * rs * g[t + 256] + beta[t + 256];
}

// ======================= host orchestration =======================
extern "C" void kernel_launch(void* const* d_in, const int* in_sizes, int n_in,
                              void* d_out, int out_size, void* d_ws, size_t ws_size,
                              hipStream_t stream) {
  const int*   tokens  = (const int*)  d_in[0];
  const int*   vlens   = (const int*)  d_in[1];
  const float* emb     = (const float*)d_in[2];
  const float* pos     = (const float*)d_in[3];
  const float* w_q     = (const float*)d_in[4];
  const float* w_k     = (const float*)d_in[5];
  const float* w_v     = (const float*)d_in[6];
  const float* w_o     = (const float*)d_in[7];
  const float* ln1_g   = (const float*)d_in[8];
  const float* ln1_b   = (const float*)d_in[9];
  const float* ffn_w1  = (const float*)d_in[10];
  const float* ffn_b1  = (const float*)d_in[11];
  const float* ffn_w2  = (const float*)d_in[12];
  const float* ffn_b2  = (const float*)d_in[13];
  const float* ln2_g   = (const float*)d_in[14];
  const float* ln2_b   = (const float*)d_in[15];
  float* outp = (float*)d_out;

  char* base = (char*)d_ws;
  size_t off = 0;
  auto take = [&](size_t bytes) -> char* {
    char* p = base + off;
    off = (off + bytes + 255) & ~(size_t)255;
    return p;
  };
  float*  x     = (float*)take((size_t)MBS * NDM * 4);
  float*  proj  = (float*)take((size_t)MBS * NDM * 4);
  float*  oacc  = (float*)take((size_t)MBS * NDM * 4);
  float*  fbuf  = (float*)take((size_t)MBS * NHID * 4);
  bf16_t* xb    = (bf16_t*)take((size_t)MBS * NHID * 2);
  bf16_t* qb    = (bf16_t*)take((size_t)NBH * NS * NDH * 2);
  bf16_t* kbb   = (bf16_t*)take((size_t)NBH * NS * NDH * 2);
  bf16_t* vtb   = (bf16_t*)take((size_t)NBH * NDH * S2 * 2);
  bf16_t* attnb = (bf16_t*)take((size_t)NBH * NS * S2 * 2);
  bf16_t* wqb   = (bf16_t*)take((size_t)NL * NDM * NDM * 2);
  bf16_t* wkb   = (bf16_t*)take((size_t)NL * NDM * NDM * 2);
  bf16_t* wvb   = (bf16_t*)take((size_t)NL * NDM * NDM * 2);
  bf16_t* wob   = (bf16_t*)take((size_t)NL * NDM * NDM * 2);
  bf16_t* w1b   = (bf16_t*)take((size_t)NL * NHID * NDM * 2);
  bf16_t* w2b   = (bf16_t*)take((size_t)NL * NDM * NHID * 2);
  if (off > ws_size) return;

  auto cvt = [&](const float* src, bf16_t* dst, int n) {
    cvt_kernel<<<(n + 255) / 256, 256, 0, stream>>>(src, dst, n);
  };

  // per-call weight conversions
  cvt(w_q,    wqb, NL * NDM * NDM);
  cvt(w_k,    wkb, NL * NDM * NDM);
  cvt(w_v,    wvb, NL * NDM * NDM);
  cvt(w_o,    wob, NL * NDM * NDM);
  cvt(ffn_w1, w1b, NL * NHID * NDM);
  cvt(ffn_w2, w2b, NL * NDM * NHID);

  embed_kernel<<<(MBS * NDM) / 256, 256, 0, stream>>>(tokens, emb, pos, x);

  const int GM = (MBS + BM - 1) / BM;                  // 63
  dim3 gAttn(NBH, (NS + 15) / 16);                     // 64 x 63

  for (int l = 0; l < NL; l++) {
    const bf16_t* wq = wqb + (size_t)l * NDM * NDM;
    const bf16_t* wk = wkb + (size_t)l * NDM * NDM;
    const bf16_t* wv = wvb + (size_t)l * NDM * NDM;
    const bf16_t* wo = wob + (size_t)l * NDM * NDM;
    const bf16_t* w1 = w1b + (size_t)l * NHID * NDM;
    const bf16_t* w2 = w2b + (size_t)l * NDM * NHID;

    // ---- QKV projections ----
    cvt(x, xb, MBS * NDM);
    gemm_kernel<<<dim3(GM, NDM / BN), 256, 0, stream>>>(xb, NDM, wq, NDM, proj, NDM, MBS, NDM, NDM, nullptr, 0);
    head_reshape_kernel<<<(MBS * NDM) / 256, 256, 0, stream>>>(proj, qb);
    gemm_kernel<<<dim3(GM, NDM / BN), 256, 0, stream>>>(xb, NDM, wk, NDM, proj, NDM, MBS, NDM, NDM, nullptr, 0);
    head_reshape_kernel<<<(MBS * NDM) / 256, 256, 0, stream>>>(proj, kbb);
    gemm_kernel<<<dim3(GM, NDM / BN), 256, 0, stream>>>(xb, NDM, wv, NDM, proj, NDM, MBS, NDM, NDM, nullptr, 0);
    vt_reshape_kernel<<<(NBH * NDH * S2) / 256, 256, 0, stream>>>(proj, vtb);

    // ---- attention ----
    scores_softmax_kernel<<<gAttn, 256, 0, stream>>>(qb, kbb, attnb, vlens);
    attnv_kernel<<<gAttn, 128, 0, stream>>>(attnb, vtb, oacc);

    // ---- output projection + LN1 ----
    cvt(oacc, xb, MBS * NDM);
    gemm_kernel<<<dim3(GM, NDM / BN), 256, 0, stream>>>(xb, NDM, wo, NDM, proj, NDM, MBS, NDM, NDM, nullptr, 0);
    ln_kernel<<<MBS, 256, 0, stream>>>(x, proj, ln1_g + (size_t)l * NDM, ln1_b + (size_t)l * NDM, x);

    // ---- FFN ----
    cvt(x, xb, MBS * NDM);
    gemm_kernel<<<dim3(GM, NHID / BN), 256, 0, stream>>>(xb, NDM, w1, NDM, fbuf, NHID, MBS, NHID, NDM,
                                                         ffn_b1 + (size_t)l * NHID, 1);
    cvt(fbuf, xb, MBS * NHID);
    gemm_kernel<<<dim3(GM, NDM / BN), 256, 0, stream>>>(xb, NHID, w2, NHID, proj, NDM, MBS, NDM, NHID,
                                                        ffn_b2 + (size_t)l * NDM, 0);
    float* lnout = (l == NL - 1) ? outp : x;
    ln_kernel<<<MBS, 256, 0, stream>>>(x, proj, ln2_g + (size_t)l * NDM, ln2_b + (size_t)l * NDM, lnout);
  }
  (void)in_sizes; (void)n_in; (void)out_size;
}
